// SegSmall_46497315947168
// MI455X (gfx1250) — compile-verified, matrix operand order checked by reading, not run on hardware
//
#include <hip/hip_runtime.h>
#include <hip/hip_bf16.h>

// ConvPoint segmentation forward for MI455X (gfx1250).
// Heavy math (MLP over B*M*K rows, per-point [16, C*16]x[C*16,cout] GEMM)
// runs on v_wmma_f32_16x16x32_f16 with f32 accumulate; everything fits in the
// 192MB L2 so the kernel is compute-bound and f16 WMMA is the right ceiling.
// KNN / einsum / batchnorm stay VALU (tiny shapes).

typedef __attribute__((ext_vector_type(16))) _Float16 v16h;
typedef __attribute__((ext_vector_type(8)))  float    v8f;
typedef _Float16 h16;

struct alignas(16) H8 { h16 v[8]; };

static __device__ __forceinline__ v8f wmma_f16(v16h a, v16h b, v8f c) {
  // D = A(16x32 f16) * B(32x16 f16) + C(16x16 f32)
  return __builtin_amdgcn_wmma_f32_16x16x32_f16(
      /*neg_a=*/false, a, /*neg_b=*/false, b,
      /*c_mod=*/(short)0, c, /*reuse_a=*/false, /*reuse_b=*/false);
}

// A fragment (16x32, f16) from an LDS matrix h[row][ld].
// Lane L<16 holds row L, K = kbase+[0..7] and kbase+[16..23];
// lane L>=16 holds row L-16, K = kbase+8+[0..7] and kbase+24+[0..7].
static __device__ __forceinline__ v16h load_a_frag(const h16* h, int ld,
                                                   int rowbase, int kbase) {
  int lane = threadIdx.x & 31;
  const h16* p = h + (size_t)(rowbase + (lane & 15)) * ld
                   + kbase + ((lane >> 4) << 3);
  H8 lo = *(const H8*)p;
  H8 hi = *(const H8*)(p + 16);
  v16h a;
#pragma unroll
  for (int i = 0; i < 8; ++i) { a[i] = lo.v[i]; a[8 + i] = hi.v[i]; }
  return a;
}

// B fragment from weights pre-packed as [frag][lane][16 halves] (contiguous).
static __device__ __forceinline__ v16h load_b_frag(const h16* __restrict__ base,
                                                   int frag) {
  int lane = threadIdx.x & 31;
  const h16* p = base + ((size_t)frag * 32 + lane) * 16;
  H8 lo = *(const H8*)p;
  H8 hi = *(const H8*)(p + 8);
  v16h b;
#pragma unroll
  for (int i = 0; i < 8; ++i) { b[i] = lo.v[i]; b[8 + i] = hi.v[i]; }
  return b;
}

// Store C fragment (16x16 f32) to LDS f16 with bias + relu.
// C layout: VGPR r, lanes 0-15 -> (M=r, N=lane); lanes 16-31 -> (M=r+8, N=lane-16).
static __device__ __forceinline__ void store_c_bias_relu(
    h16* dst, int ld, int rowbase, const float* __restrict__ bias, int nbase,
    v8f c, bool zero_hi) {
  int lane = threadIdx.x & 31;
  int col  = lane & 15;
  int mofs = (lane >> 4) << 3;
  float bv = bias[nbase + col];
#pragma unroll
  for (int r = 0; r < 8; ++r) {
    float v = c[r] + bv;
    v = v > 0.f ? v : 0.f;
    h16* q = dst + (size_t)(rowbase + r + mofs) * ld + nbase + col;
    q[0] = (h16)v;
    if (zero_hi) q[16] = (h16)0.f;
  }
}

// ---------------------------------------------------------------------------
// Weight packing: W[Kd,Nd] f32 row-major -> f16 B-fragments, zero padded.
// frag = nt*KC + kc ; element j of lane l maps to
//   n = nt*16 + (l&15),  k = kc*32 + ((j>=8)?16:0) + ((l>>4)<<3) + (j&7)
// ---------------------------------------------------------------------------
__global__ void pack_wB(const float* __restrict__ W, int Kd, int Nd,
                        h16* __restrict__ out) {
  int KC = (Kd + 31) / 32;
  int NT = (Nd + 15) / 16;
  int total = KC * NT * 32;
  int t = blockIdx.x * blockDim.x + threadIdx.x;
  if (t >= total) return;
  int lane = t & 31;
  int frag = t >> 5;
  int kc = frag % KC;
  int nt = frag / KC;
  int n = nt * 16 + (lane & 15);
  int kbase = kc * 32 + ((lane >> 4) << 3);
#pragma unroll
  for (int j = 0; j < 16; ++j) {
    int k = kbase + ((j >= 8) ? 16 : 0) + (j & 7);
    float v = (k < Kd && n < Nd) ? W[(size_t)k * Nd + n] : 0.f;
    out[(size_t)t * 16 + j] = (h16)v;
  }
}

// ---------------------------------------------------------------------------
// KNN: one thread per query, LDS-tiled source points, register top-K ladder.
// ---------------------------------------------------------------------------
template <int KK>
__global__ void knn_kernel(const float* __restrict__ pts, int pstride,
                           int Ns, int M, int* __restrict__ idx) {
  __shared__ float tx[256], ty[256], tz[256];
  int b = blockIdx.y;
  int q = blockIdx.x * blockDim.x + threadIdx.x;
  bool valid = q < M;
  const float* pb = pts + (size_t)b * pstride;
  float qx = 0.f, qy = 0.f, qz = 0.f;
  if (valid) { qx = pb[q * 3]; qy = pb[q * 3 + 1]; qz = pb[q * 3 + 2]; }
  float bd[KK];
  int bi[KK];
#pragma unroll
  for (int i = 0; i < KK; ++i) { bd[i] = 3.4e38f; bi[i] = 0; }
  for (int t0 = 0; t0 < Ns; t0 += 256) {
    __syncthreads();
    int j = t0 + threadIdx.x;
    if (j < Ns) {
      tx[threadIdx.x] = pb[j * 3];
      ty[threadIdx.x] = pb[j * 3 + 1];
      tz[threadIdx.x] = pb[j * 3 + 2];
    }
    __syncthreads();
    int cnt = Ns - t0; if (cnt > 256) cnt = 256;
    if (valid) {
      for (int jj = 0; jj < cnt; ++jj) {
        float dx = qx - tx[jj], dy = qy - ty[jj], dz = qz - tz[jj];
        float dist = dx * dx + dy * dy + dz * dz;
        int ii = t0 + jj;
#pragma unroll
        for (int p = 0; p < KK; ++p) {       // compare/swap insertion ladder
          if (dist < bd[p]) {
            float td = bd[p]; bd[p] = dist; dist = td;
            int   ti = bi[p]; bi[p] = ii;   ii   = ti;
          }
        }
      }
    }
  }
  if (valid) {
#pragma unroll
    for (int k = 0; k < KK; ++k)
      idx[((size_t)b * M + q) * KK + k] = bi[k];
  }
}

// ---------------------------------------------------------------------------
// PtConv core: one workgroup = one (batch, 16-query tile). 8 waves.
//  1. build d[16K,64] (f16, LDS)
//  2. per-wave WMMA MLP: d->h1(32)->h2(16)->h3(16), relu+bias between
//  3. VALU einsum: F[16, C*16] = sum_k fg[q,k,c]*h3[q,k,n]  (LDS f16)
//  4. WMMA GEMM: out[16,cout] = F @ packed_weight, scaled by 1/K
// ---------------------------------------------------------------------------
__global__ __launch_bounds__(256, 1)
void ptconv_kernel(const float* __restrict__ feat, int C,
                   const float* __restrict__ pts, int pstride,
                   int Ns, int M, int K,
                   const int* __restrict__ idx,
                   const float* __restrict__ centers,
                   const h16* __restrict__ w1p, const float* __restrict__ b1,
                   const h16* __restrict__ w2p, const float* __restrict__ b2,
                   const h16* __restrict__ w3p, const float* __restrict__ b3,
                   const h16* __restrict__ wfp, int cout,
                   float* __restrict__ out) {
  extern __shared__ char smem[];
  int b = blockIdx.y;
  int mbase = blockIdx.x * 16;
  int mrem = M - mbase; if (mrem > 16) mrem = 16;
  int R  = 16 * K;
  int CK = ((C * 16 + 31) / 32) * 32;
  int tid = threadIdx.x;
  int wave = tid >> 5;

  // LDS carve (all section sizes are multiples of 16 bytes)
  float* sq    = (float*)smem;          // [16][3]
  float* scent = sq + 48;               // [48]
  int*   snidx = (int*)(scent + 48);    // [16][K]
  float* spg   = (float*)(snidx + 16 * K); // [R][3]
  char*  cp    = (char*)(spg + (size_t)R * 3);
  h16* sd  = (h16*)cp;  cp += (size_t)R * 64 * 2;
  h16* sh1 = (h16*)cp;  cp += (size_t)R * 32 * 2;
  h16* sh2 = (h16*)cp;  cp += (size_t)R * 32 * 2;
  h16* sh3 = (h16*)cp;  cp += (size_t)R * 16 * 2;
  h16* sF  = (h16*)cp;                  // [16][CK]

  const float* pb = pts + (size_t)b * pstride;

  if (tid < 48) {
    scent[tid] = centers[tid];
    int q = tid / 3, d = tid % 3;
    int qq = mbase + q; if (qq > M - 1) qq = M - 1;
    sq[tid] = pb[qq * 3 + d];
  }
  for (int t = tid; t < 16 * K; t += 256) {
    int q = t / K;
    int qq = mbase + q; if (qq > M - 1) qq = M - 1;
    snidx[t] = idx[((size_t)b * M + qq) * K + (t % K)];
  }
  __syncthreads();

  for (int r = tid; r < R; r += 256) {
    int q = r / K;
    int n = snidx[r];
    __builtin_prefetch(&feat[((size_t)b * Ns + n) * C], 0, 1);
    spg[r * 3 + 0] = pb[n * 3 + 0] - sq[q * 3 + 0];
    spg[r * 3 + 1] = pb[n * 3 + 1] - sq[q * 3 + 1];
    spg[r * 3 + 2] = pb[n * 3 + 2] - sq[q * 3 + 2];
  }
  __syncthreads();

  // d[r][j] = pg[r][j/16] - centers[j], padded 48..63 with zeros
  for (int t = tid; t < R * 64; t += 256) {
    int r = t >> 6, j = t & 63;
    float v = (j < 48) ? (spg[r * 3 + (j >> 4)] - scent[j]) : 0.f;
    sd[t] = (h16)v;
  }
  __syncthreads();

  // WMMA MLP, one 16-row tile per wave iteration (wave-uniform control flow).
  for (int rt = wave; rt < K; rt += 8) {
    int rowbase = rt * 16;
    v8f c0 = {}, c1 = {};
#pragma unroll
    for (int kc = 0; kc < 2; ++kc) {
      v16h a = load_a_frag(sd, 64, rowbase, kc * 32);
      c0 = wmma_f16(a, load_b_frag(w1p, 0 * 2 + kc), c0);
      c1 = wmma_f16(a, load_b_frag(w1p, 1 * 2 + kc), c1);
    }
    store_c_bias_relu(sh1, 32, rowbase, b1, 0,  c0, false);
    store_c_bias_relu(sh1, 32, rowbase, b1, 16, c1, false);

    v16h a2 = load_a_frag(sh1, 32, rowbase, 0);
    v8f c2 = {};
    c2 = wmma_f16(a2, load_b_frag(w2p, 0), c2);
    store_c_bias_relu(sh2, 32, rowbase, b2, 0, c2, true);   // zero k=16..31 pad

    v16h a3 = load_a_frag(sh2, 32, rowbase, 0);
    v8f c3 = {};
    c3 = wmma_f16(a3, load_b_frag(w3p, 0), c3);
    store_c_bias_relu(sh3, 16, rowbase, b3, 0, c3, false);
  }
  __syncthreads();

  // F[q][c*16+n] = sum_k fg[q,k,c] * h3[q*K+k][n]
  for (int t = tid; t < 16 * C; t += 256) {
    int q = t / C, c = t % C;
    float acc[16];
#pragma unroll
    for (int j = 0; j < 16; ++j) acc[j] = 0.f;
    for (int k = 0; k < K; ++k) {
      int n = snidx[q * K + k];
      float fgv = feat[((size_t)b * Ns + n) * C + c];
      const h16* hp = sh3 + (size_t)(q * K + k) * 16;
#pragma unroll
      for (int j = 0; j < 16; ++j) acc[j] += fgv * (float)hp[j];
    }
    h16* Fp = sF + (size_t)q * CK + c * 16;
#pragma unroll
    for (int j = 0; j < 16; ++j) Fp[j] = (h16)acc[j];
  }
  int padc = CK - C * 16;
  for (int t = tid; t < 16 * padc; t += 256) {
    int q = t / padc, j = t % padc;
    sF[(size_t)q * CK + C * 16 + j] = (h16)0.f;
  }
  __syncthreads();

  // Final GEMM: each wave owns one 16-wide output tile (wave-uniform).
  int ntiles = cout >> 4;
  int KC2 = CK >> 5;
  float invK = 1.f / (float)K;
  for (int nt = wave; nt < ntiles; nt += 8) {
    v8f c = {};
    for (int kc = 0; kc < KC2; ++kc) {
      v16h a = load_a_frag(sF, CK, 0, kc * 32);
      c = wmma_f16(a, load_b_frag(wfp, nt * KC2 + kc), c);
    }
    int lane = tid & 31;
    int col  = lane & 15;
    int mofs = (lane >> 4) << 3;
#pragma unroll
    for (int r = 0; r < 8; ++r) {
      int ml = r + mofs;
      if (ml < mrem)
        out[((size_t)b * M + mbase + ml) * cout + nt * 16 + col] = c[r] * invK;
    }
  }
}

// ---------------------------------------------------------------------------
// BatchNorm (training-mode batch stats) + relu, concat copy, final FC.
// ---------------------------------------------------------------------------
__global__ void bn_stats_kernel(const float* __restrict__ x, int Npts, int Cc,
                                float* __restrict__ mv) {
  __shared__ float ss[256], ss2[256];
  int c = blockIdx.x;
  float s = 0.f, s2 = 0.f;
  for (int i = threadIdx.x; i < Npts; i += 256) {
    float v = x[(size_t)i * Cc + c];
    s += v; s2 += v * v;
  }
  ss[threadIdx.x] = s; ss2[threadIdx.x] = s2;
  __syncthreads();
  for (int o = 128; o > 0; o >>= 1) {
    if (threadIdx.x < o) {
      ss[threadIdx.x]  += ss[threadIdx.x + o];
      ss2[threadIdx.x] += ss2[threadIdx.x + o];
    }
    __syncthreads();
  }
  if (threadIdx.x == 0) {
    float m = ss[0] / (float)Npts;
    mv[c] = m;
    mv[Cc + c] = ss2[0] / (float)Npts - m * m;
  }
}

__global__ void bn_apply_kernel(const float* __restrict__ x, int Npts, int Cc,
                                const float* __restrict__ mv,
                                const float* __restrict__ g,
                                const float* __restrict__ bb,
                                float* __restrict__ dst, int dld, int doff) {
  size_t t = (size_t)blockIdx.x * blockDim.x + threadIdx.x;
  if (t >= (size_t)Npts * Cc) return;
  int c = (int)(t % Cc);
  size_t r = t / Cc;
  float y = (x[t] - mv[c]) * rsqrtf(mv[Cc + c] + 1e-5f) * g[c] + bb[c];
  dst[r * dld + doff + c] = y > 0.f ? y : 0.f;
}

__global__ void copy_ch_kernel(const float* __restrict__ src, int Npts, int Cc,
                               float* __restrict__ dst, int dld, int doff) {
  size_t t = (size_t)blockIdx.x * blockDim.x + threadIdx.x;
  if (t >= (size_t)Npts * Cc) return;
  int c = (int)(t % Cc);
  size_t r = t / Cc;
  dst[r * dld + doff + c] = src[t];
}

__global__ void fc_kernel(const float* __restrict__ x, int Npts,
                          const float* __restrict__ W,
                          const float* __restrict__ bias,
                          float* __restrict__ out) {
  size_t t = (size_t)blockIdx.x * blockDim.x + threadIdx.x;
  if (t >= (size_t)Npts * 20) return;
  int o = (int)(t % 20);
  size_t r = t / 20;
  float a = bias[o];
  const float* xr = x + r * 48;
#pragma unroll
  for (int c = 0; c < 48; ++c) a += xr[c] * W[c * 20 + o];
  out[t] = a;
}

// ---------------------------------------------------------------------------
// Host orchestration. Input layout (insertion order of setup_inputs dicts):
//   d_in[0]=x, d_in[1]=input_pts, then per layer i (10 layers, 10 arrays):
//   weight, centers, w1, b1, w2, b2, w3, b3, bn_g, bn_b ; then fc_w, fc_b.
// ---------------------------------------------------------------------------
extern "C" void kernel_launch(void* const* d_in, const int* in_sizes, int n_in,
                              void* d_out, int out_size, void* d_ws, size_t ws_size,
                              hipStream_t stream) {
  (void)in_sizes; (void)n_in; (void)out_size; (void)ws_size;
  const float* x   = (const float*)d_in[0];
  const float* pts = (const float*)d_in[1];
  auto P = [&](int layer, int j) -> const float* {
    return (const float*)d_in[2 + layer * 10 + j];
  };
  const float* fc_w = (const float*)d_in[102];
  const float* fc_b = (const float*)d_in[103];

  // bump allocator over workspace (~39 MB total)
  char* ws = (char*)d_ws;
  size_t cur = 0;
  auto alloc = [&](size_t bytes) -> void* {
    cur = (cur + 255) & ~(size_t)255;
    void* p = ws + cur;
    cur += bytes;
    return p;
  };

  float* tmp = (float*)alloc(8ull * 8192 * 48 * 4);
  float* x2  = (float*)alloc(8ull * 1024 * 48 * 4);
  float* x3  = (float*)alloc(8ull * 256 * 48 * 4);
  float* x4  = (float*)alloc(8ull * 64 * 96 * 4);
  float* x5  = (float*)alloc(8ull * 16 * 96 * 4);
  float* x6  = (float*)alloc(8ull * 8 * 96 * 4);
  float* c5d = (float*)alloc(8ull * 16 * 192 * 4);
  float* c4d = (float*)alloc(8ull * 64 * 192 * 4);
  float* c3d = (float*)alloc(8ull * 256 * 96 * 4);
  float* c2d = (float*)alloc(8ull * 1024 * 96 * 4);
  float* x1d = (float*)alloc(8ull * 8192 * 48 * 4);
  int*   idxb = (int*)alloc(8ull * 8192 * 16 * 4);   // reused per layer
  float* mv   = (float*)alloc(256 * 4);

  static const int Cs[10]    = {3, 48, 48, 96, 96, 96, 192, 192, 96, 96};
  static const int Nss[10]   = {8192, 1024, 256, 64, 16, 8, 16, 64, 256, 1024};
  static const int Ms[10]    = {1024, 256, 64, 16, 8, 16, 64, 256, 1024, 8192};
  static const int Ks[10]    = {16, 16, 8, 8, 4, 4, 4, 4, 8, 8};
  static const int couts[10] = {48, 48, 96, 96, 96, 96, 96, 48, 48, 48};
  const float* feats[10] = {x, x2, x3, x4, x5, x6, c5d, c4d, c3d, c2d};
  float*       dsts[10]  = {x2, x3, x4, x5, x6, c5d, c4d, c3d, c2d, x1d};
  static const int dlds[10] = {48, 48, 96, 96, 96, 192, 192, 96, 96, 48};

  // pack all weights into f16 B-fragments
  auto pack = [&](const float* W, int Kd, int Nd) -> h16* {
    int KC = (Kd + 31) / 32, NT = (Nd + 15) / 16;
    int total = KC * NT * 32;
    h16* outp = (h16*)alloc((size_t)total * 16 * 2);
    pack_wB<<<(total + 255) / 256, 256, 0, stream>>>(W, Kd, Nd, outp);
    return outp;
  };
  h16 *w1p[10], *w2p[10], *w3p[10], *wfp[10];
  for (int i = 0; i < 10; ++i) {
    wfp[i] = pack(P(i, 0), Cs[i] * 16, couts[i]);
    w1p[i] = pack(P(i, 2), 48, 32);
    w2p[i] = pack(P(i, 4), 32, 16);
    w3p[i] = pack(P(i, 6), 16, 16);
  }

  auto smem_bytes = [](int C, int K) -> size_t {
    int R = 16 * K;
    int CK = ((C * 16 + 31) / 32) * 32;
    size_t s = 48 * 4 + 48 * 4 + (size_t)16 * K * 4 + (size_t)R * 3 * 4;
    s += (size_t)R * 64 * 2 + (size_t)R * 32 * 2 + (size_t)R * 32 * 2 +
         (size_t)R * 16 * 2;
    s += (size_t)16 * CK * 2;
    return s;
  };

  const int PST = 8192 * 3;
  for (int i = 0; i < 10; ++i) {
    int C = Cs[i], Ns = Nss[i], Mq = Ms[i], K = Ks[i], co = couts[i];

    dim3 kgrid((Mq + 255) / 256, 8);
    switch (K) {
      case 4:  knn_kernel<4><<<kgrid, 256, 0, stream>>>(pts, PST, Ns, Mq, idxb); break;
      case 8:  knn_kernel<8><<<kgrid, 256, 0, stream>>>(pts, PST, Ns, Mq, idxb); break;
      default: knn_kernel<16><<<kgrid, 256, 0, stream>>>(pts, PST, Ns, Mq, idxb); break;
    }

    dim3 pgrid((Mq + 15) / 16, 8);
    ptconv_kernel<<<pgrid, 256, smem_bytes(C, K), stream>>>(
        feats[i], C, pts, PST, Ns, Mq, K, idxb, P(i, 1),
        w1p[i], P(i, 3), w2p[i], P(i, 5), w3p[i], P(i, 7),
        wfp[i], co, tmp);

    int Npts = 8 * Mq;
    bn_stats_kernel<<<co, 256, 0, stream>>>(tmp, Npts, co, mv);
    size_t tot = (size_t)Npts * co;
    bn_apply_kernel<<<(unsigned)((tot + 255) / 256), 256, 0, stream>>>(
        tmp, Npts, co, mv, P(i, 8), P(i, 9), dsts[i], dlds[i], 0);

    // skip-connection concats into the decoder buffers
    if (i == 5) copy_ch_kernel<<<(8 * 16 * 96 + 255) / 256, 256, 0, stream>>>(
        x5, 8 * 16, 96, c5d, 192, 96);
    if (i == 6) copy_ch_kernel<<<(8 * 64 * 96 + 255) / 256, 256, 0, stream>>>(
        x4, 8 * 64, 96, c4d, 192, 96);
    if (i == 7) copy_ch_kernel<<<(8 * 256 * 48 + 255) / 256, 256, 0, stream>>>(
        x3, 8 * 256, 48, c3d, 96, 48);
    if (i == 8) copy_ch_kernel<<<(8 * 1024 * 48 + 255) / 256, 256, 0, stream>>>(
        x2, 8 * 1024, 48, c2d, 96, 48);
  }

  size_t tot = 8ull * 8192 * 20;
  fc_kernel<<<(unsigned)((tot + 255) / 256), 256, 0, stream>>>(
      x1d, 8 * 8192, fc_w, fc_b, (float*)d_out);
}
